// MergeModuleTemporal_63385127354790
// MI455X (gfx1250) — compile-verified, matrix operand order checked by reading, not run on hardware
//
#include <hip/hip_runtime.h>
#include <math.h>

// ---------------------------------------------------------------------------
// MergeModuleTemporal on MI455X (gfx1250, wave32)
//
// Per round (x4):
//   1) convert emb fp32 -> bf16 (ws)
//   2) sim = E E^T / 1024 via v_wmma_f32_16x16x32_bf16
//      - upper-triangular 32x32 super-tiles, 2x2 register blocking per wave
//      - operand fragments reused 2x -> 1KB L2 traffic per WMMA
//   3) sequential greedy scan (1 workgroup, wave32 shuffle argmax)
//   4) fuse rows (min(a+b,1)), 5) zero consumed rows + kill alive bits
// ---------------------------------------------------------------------------

typedef __attribute__((ext_vector_type(16))) __bf16 v16bf;
typedef __attribute__((ext_vector_type(8)))  float  v8f;

#define N_ROWS 4096
#define DIM    1024
#define NT2    (N_ROWS / 32)               // 128 super-tiles per side
#define TRI2_TILES ((NT2 * (NT2 + 1)) / 2) // 8256 upper-triangular super-tiles
#define WAVES_PER_BLOCK 8
#define THRV   0.25f
#define ROUNDS 4

__device__ __forceinline__ int tri2_off(int r) {  // first super-tile of row r
    return r * NT2 - (r * (r - 1)) / 2;
}

// ---------------- init: copy input emb -> d_out working buffer --------------
__global__ void k_copy(const float* __restrict__ in, float* __restrict__ out) {
    size_t idx = (size_t)blockIdx.x * blockDim.x + threadIdx.x;   // 1M threads x float4
    ((float4*)out)[idx] = ((const float4*)in)[idx];
}

__global__ void k_meta(int* __restrict__ aliveInt, int* __restrict__ flags) {
    int t = threadIdx.x;
    for (int i = t; i < N_ROWS; i += blockDim.x) aliveInt[i] = 1;
    if (t == 0) flags[0] = 0;   // done flag
}

// ---------------- fp32 -> bf16 (RNE) ----------------------------------------
__global__ void k_convert(const float* __restrict__ emb,
                          unsigned short* __restrict__ eb,
                          const int* __restrict__ flags) {
    if (flags[0]) return;
    size_t base = ((size_t)blockIdx.x * blockDim.x + threadIdx.x) * 4;
    #pragma unroll
    for (int k = 0; k < 4; k++) {
        unsigned int u = __float_as_uint(emb[base + k]);
        unsigned int r = (u + 0x7FFFu + ((u >> 16) & 1u)) >> 16;
        eb[base + k] = (unsigned short)r;
    }
}

// ---------------- Gram matrix: one wave per 32x32 super-tile ----------------
union Pk { uint4 q[2]; v16bf v; };

__device__ __forceinline__ void store_tile(float* __restrict__ sim,
                                           int i0, int j0, int half, int m,
                                           v8f c, float s) {
    // D layout: VGPR r -> lanes 0-15: (M=r, N=lane); lanes 16-31: (M=8+r, N=lane-16)
    const int col = j0 + m;
    const int rowb = i0 + half * 8;
    #pragma unroll
    for (int r = 0; r < 8; r++)
        sim[(size_t)(rowb + r) * N_ROWS + col] = c[r] * s;
}

__global__ void k_gemm(const unsigned short* __restrict__ eb,
                       float* __restrict__ sim,
                       const int* __restrict__ flags) {
    if (flags[0]) return;
    int wave = blockIdx.x * WAVES_PER_BLOCK + (threadIdx.x >> 5);
    if (wave >= TRI2_TILES) return;                     // wave-uniform exit

    // decode linear upper-tri index -> (ti, tj) over 128x128 super-tiles
    int ti = (int)((257.0f - sqrtf(257.0f * 257.0f - 8.0f * (float)wave)) * 0.5f);
    if (ti > NT2 - 1) ti = NT2 - 1;
    if (ti < 0) ti = 0;
    while (ti > 0 && wave < tri2_off(ti)) --ti;
    while (ti < NT2 - 1 && wave >= tri2_off(ti + 1)) ++ti;
    int tj = ti + (wave - tri2_off(ti));

    const int i0 = ti * 32, j0 = tj * 32;
    const int lane = threadIdx.x & 31;
    const int half = lane >> 4;          // K sub-phase per ISA A/B 16-bit layout
    const int m    = lane & 15;          // row (A/B fragments) / col (D)

    // A fragments: rows (i0+m), (i0+16+m); B fragments: rows (j0+m), (j0+16+m).
    // Gram matrix: B = E^T, so B operand == A-pattern on the j-tile rows.
    const unsigned short* rA0 = eb + (size_t)(i0 + m) * DIM + half * 8;
    const unsigned short* rA1 = rA0 + (size_t)16 * DIM;
    const unsigned short* rB0 = eb + (size_t)(j0 + m) * DIM + half * 8;
    const unsigned short* rB1 = rB0 + (size_t)16 * DIM;

    v8f c00 = {}, c01 = {}, c10 = {}, c11 = {};
    for (int k0 = 0; k0 < DIM; k0 += 32) {
        Pk a0, a1, b0, b1;
        a0.q[0] = *(const uint4*)(rA0 + k0);
        a0.q[1] = *(const uint4*)(rA0 + k0 + 16);
        a1.q[0] = *(const uint4*)(rA1 + k0);
        a1.q[1] = *(const uint4*)(rA1 + k0 + 16);
        b0.q[0] = *(const uint4*)(rB0 + k0);
        b0.q[1] = *(const uint4*)(rB0 + k0 + 16);
        b1.q[0] = *(const uint4*)(rB1 + k0);
        b1.q[1] = *(const uint4*)(rB1 + k0 + 16);
        if (k0 + 64 < DIM) {                            // global_prefetch_b8
            __builtin_prefetch(rA0 + k0 + 64, 0, 1);
            __builtin_prefetch(rA1 + k0 + 64, 0, 1);
            __builtin_prefetch(rB0 + k0 + 64, 0, 1);
            __builtin_prefetch(rB1 + k0 + 64, 0, 1);
        }
        c00 = __builtin_amdgcn_wmma_f32_16x16x32_bf16(false, a0.v, false, b0.v, (short)0, c00, false, false);
        c01 = __builtin_amdgcn_wmma_f32_16x16x32_bf16(false, a0.v, false, b1.v, (short)0, c01, false, false);
        c10 = __builtin_amdgcn_wmma_f32_16x16x32_bf16(false, a1.v, false, b0.v, (short)0, c10, false, false);
        c11 = __builtin_amdgcn_wmma_f32_16x16x32_bf16(false, a1.v, false, b1.v, (short)0, c11, false, false);
    }

    const float scale = 1.0f / (float)DIM;
    store_tile(sim, i0,      j0,      half, m, c00, scale);
    store_tile(sim, i0,      j0 + 16, half, m, c01, scale);
    store_tile(sim, i0 + 16, j0,      half, m, c10, scale);  // j<i part unused, harmless
    store_tile(sim, i0 + 16, j0 + 16, half, m, c11, scale);
}

// ---------------- sequential greedy scan (single workgroup) -----------------
__global__ void __launch_bounds__(1024)
k_scan(const float* __restrict__ sim, const int* __restrict__ aliveInt,
       int* __restrict__ partner, int* __restrict__ consumed,
       int* __restrict__ flags) {
    __shared__ unsigned int aliveB[128], mergedB[128];
    __shared__ float redV[32];
    __shared__ int   redI[32];
    __shared__ int   sOcc;
    const int tid = threadIdx.x;

    for (int i = tid; i < N_ROWS; i += 1024) { partner[i] = -1; consumed[i] = 0; }
    if (tid < 128) {
        unsigned int w = 0;
        for (int b = 0; b < 32; b++) w |= (aliveInt[tid * 32 + b] ? 1u : 0u) << b;
        aliveB[tid] = w; mergedB[tid] = 0u;
    }
    if (tid == 0) sOcc = 0;
    __syncthreads();
    if (flags[0]) return;        // done in a previous round (arrays already reset)

    for (int i = 0; i < N_ROWS; i++) {
        float best = -1e30f; int bj = 0x7fffffff;
        for (int j = i + 1 + tid; j < N_ROWS; j += 1024) {
            bool ok = ((aliveB[j >> 5] >> (j & 31)) & 1u) &&
                      !((mergedB[j >> 5] >> (j & 31)) & 1u);
            if (ok) {
                float s = sim[(size_t)i * N_ROWS + j];
                if (s > best) { best = s; bj = j; }
            }
        }
        #pragma unroll
        for (int off = 16; off; off >>= 1) {   // wave32 reduce, prefer lower j on ties
            float ov = __shfl_xor(best, off);
            int   oj = __shfl_xor(bj,   off);
            if (ov > best || (ov == best && oj < bj)) { best = ov; bj = oj; }
        }
        if ((tid & 31) == 0) { redV[tid >> 5] = best; redI[tid >> 5] = bj; }
        __syncthreads();
        if (tid < 32) {
            best = redV[tid]; bj = redI[tid];
            #pragma unroll
            for (int off = 16; off; off >>= 1) {
                float ov = __shfl_xor(best, off);
                int   oj = __shfl_xor(bj,   off);
                if (ov > best || (ov == best && oj < bj)) { best = ov; bj = oj; }
            }
            if (tid == 0) {
                bool iok = ((aliveB[i >> 5] >> (i & 31)) & 1u) &&
                           !((mergedB[i >> 5] >> (i & 31)) & 1u);
                if (iok && best >= THRV && bj < N_ROWS) {
                    partner[i] = bj;
                    consumed[bj] = 1;
                    mergedB[bj >> 5] |= 1u << (bj & 31);
                    sOcc = 1;
                }
            }
        }
        __syncthreads();
    }
    if (tid == 0) {
        int cnt = 0;
        for (int w = 0; w < 128; w++) cnt += __popc(aliveB[w] & ~mergedB[w]);
        if (!sOcc || cnt <= 1) flags[0] = 1;   // gates subsequent rounds only
    }
}

// ---------------- fuse: emb[i] = min(emb[i] + emb[p], 1) --------------------
// Safe in place: consumed rows never have partners, so rows read (p) are
// never written in this kernel.
__global__ void k_fuse(float* __restrict__ emb, const int* __restrict__ partner) {
    const int i = blockIdx.x;
    const int p = partner[i];
    if (p < 0) return;
    const float4* a = (const float4*)(emb + (size_t)i * DIM);
    const float4* b = (const float4*)(emb + (size_t)p * DIM);
    float4*       o = (float4*)(emb + (size_t)i * DIM);
    for (int t = threadIdx.x; t < DIM / 4; t += blockDim.x) {
        float4 va = a[t], vb = b[t], vo;
        vo.x = fminf(va.x + vb.x, 1.0f);
        vo.y = fminf(va.y + vb.y, 1.0f);
        vo.z = fminf(va.z + vb.z, 1.0f);
        vo.w = fminf(va.w + vb.w, 1.0f);
        o[t] = vo;
    }
}

__global__ void k_zero(float* __restrict__ emb, int* __restrict__ aliveInt,
                       const int* __restrict__ consumed) {
    const int i = blockIdx.x;
    if (!consumed[i]) return;
    if (threadIdx.x == 0) aliveInt[i] = 0;
    float4* o = (float4*)(emb + (size_t)i * DIM);
    for (int t = threadIdx.x; t < DIM / 4; t += blockDim.x)
        o[t] = make_float4(0.f, 0.f, 0.f, 0.f);
}

__global__ void k_alive_out(const int* __restrict__ aliveInt, float* __restrict__ outA) {
    int i = blockIdx.x * blockDim.x + threadIdx.x;
    if (i < N_ROWS) outA[i] = aliveInt[i] ? 1.0f : 0.0f;
}

// ---------------------------------------------------------------------------
extern "C" void kernel_launch(void* const* d_in, const int* in_sizes, int n_in,
                              void* d_out, int out_size, void* d_ws, size_t ws_size,
                              hipStream_t stream) {
    const float* inEmb = (const float*)d_in[0];
    float* outEmb   = (float*)d_out;
    float* outAlive = outEmb + (size_t)N_ROWS * DIM;

    char* ws = (char*)d_ws;
    const size_t simBytes = (size_t)N_ROWS * N_ROWS * sizeof(float);       // 64 MB
    const size_t bfBytes  = (size_t)N_ROWS * DIM * sizeof(unsigned short); // 8 MB
    float*          sim      = (float*)ws;
    unsigned short* ebf      = (unsigned short*)(ws + simBytes);
    int*            aliveInt = (int*)(ws + simBytes + bfBytes);
    int*            partner  = aliveInt + N_ROWS;
    int*            consumed = partner + N_ROWS;
    int*            flags    = consumed + N_ROWS;

    k_copy<<<4096, 256, 0, stream>>>(inEmb, outEmb);
    k_meta<<<1, 256, 0, stream>>>(aliveInt, flags);

    for (int r = 0; r < ROUNDS; r++) {
        k_convert<<<4096, 256, 0, stream>>>(outEmb, ebf, flags);
        k_gemm<<<TRI2_TILES / WAVES_PER_BLOCK, WAVES_PER_BLOCK * 32, 0, stream>>>(ebf, sim, flags);
        k_scan<<<1, 1024, 0, stream>>>(sim, aliveInt, partner, consumed, flags);
        k_fuse<<<N_ROWS, 256, 0, stream>>>(outEmb, partner);
        k_zero<<<N_ROWS, 256, 0, stream>>>(outEmb, aliveInt, consumed);
    }
    k_alive_out<<<16, 256, 0, stream>>>(aliveInt, outAlive);
}
